// RowAttentionWithPairBias_87591563034704
// MI455X (gfx1250) — compile-verified, matrix operand order checked by reading, not run on hardware
//
#include <hip/hip_runtime.h>
#include <hip/hip_bf16.h>

typedef __attribute__((ext_vector_type(16))) _Float16 v16h;
typedef __attribute__((ext_vector_type(8)))  float    v8f;

union V16H { v16h v; uint4 q[2]; };
union H4   { _Float16 h[4]; uint2 u; };

#define R_    1024
#define CIN   512
#define CHID  512
#define H_    8
#define DH_   64
#define CZ    128
#define EPSF  1e-5f
#define QSCALE 0.04419417382415922f   /* 1/sqrt(512) */

// ---------------------------------------------------------------------------
// Kernel 1: LayerNorm(m) -> f16 activations.  1 block (128 thr) per row.
// ---------------------------------------------------------------------------
__global__ void ln_m_kernel(const float* __restrict__ m,
                            const float* __restrict__ g,
                            const float* __restrict__ b,
                            _Float16* __restrict__ out) {
    const int row = blockIdx.x;
    const int tid = threadIdx.x;
    const float4 x = ((const float4*)(m + (size_t)row * CIN))[tid];

    __shared__ float red[128];
    float s = x.x + x.y + x.z + x.w;
    red[tid] = s; __syncthreads();
    #pragma unroll
    for (int o = 64; o > 0; o >>= 1) { if (tid < o) red[tid] += red[tid + o]; __syncthreads(); }
    const float mu = red[0] * (1.0f / CIN);
    __syncthreads();

    const float dx = x.x - mu, dy = x.y - mu, dz = x.z - mu, dw = x.w - mu;
    red[tid] = dx*dx + dy*dy + dz*dz + dw*dw; __syncthreads();
    #pragma unroll
    for (int o = 64; o > 0; o >>= 1) { if (tid < o) red[tid] += red[tid + o]; __syncthreads(); }
    const float rinv = rsqrtf(red[0] * (1.0f / CIN) + EPSF);

    const int c = tid * 4;
    const float4 gg = ((const float4*)g)[tid];
    const float4 bb = ((const float4*)b)[tid];
    H4 o;
    o.h[0] = (_Float16)(dx * rinv * gg.x + bb.x);
    o.h[1] = (_Float16)(dy * rinv * gg.y + bb.y);
    o.h[2] = (_Float16)(dz * rinv * gg.z + bb.z);
    o.h[3] = (_Float16)(dw * rinv * gg.w + bb.w);
    *(uint2*)(out + (size_t)row * CIN + c) = o.u;
}

// ---------------------------------------------------------------------------
// Kernel 2: transpose+convert 5 weight matrices [K][N] f32 -> [N][K] f16
// block (32,8); grid (16,16,5); 32x32 tile via LDS.
// ---------------------------------------------------------------------------
__global__ void cvt_w_kernel(const float* __restrict__ Wq, const float* __restrict__ Wk,
                             const float* __restrict__ Wv, const float* __restrict__ Wg,
                             const float* __restrict__ Wo,
                             _Float16* __restrict__ Qt, _Float16* __restrict__ Kt,
                             _Float16* __restrict__ Vt, _Float16* __restrict__ Gt,
                             _Float16* __restrict__ Ot) {
    const float* src; _Float16* dst;
    switch (blockIdx.z) {
        case 0: src = Wq; dst = Qt; break;
        case 1: src = Wk; dst = Kt; break;
        case 2: src = Wv; dst = Vt; break;
        case 3: src = Wg; dst = Gt; break;
        default: src = Wo; dst = Ot; break;
    }
    __shared__ float tile[32][33];
    const int n0 = blockIdx.x * 32, k0 = blockIdx.y * 32;
    #pragma unroll
    for (int i = 0; i < 4; ++i) {
        int k = k0 + threadIdx.y + i * 8;
        tile[threadIdx.y + i * 8][threadIdx.x] = src[(size_t)k * 512 + n0 + threadIdx.x];
    }
    __syncthreads();
    #pragma unroll
    for (int i = 0; i < 4; ++i) {
        int n = n0 + threadIdx.y + i * 8;
        dst[(size_t)n * 512 + k0 + threadIdx.x] = (_Float16)tile[threadIdx.x][threadIdx.y + i * 8];
    }
}

// ---------------------------------------------------------------------------
// Kernel 3: pair bias.  One wave per z row (i,j): LN(z[i,j,:]) @ Wz -> bias[h][i][j].
// block = 256 thr (8 waves); grid = R*R/8 blocks.
// ---------------------------------------------------------------------------
__global__ void z_bias_kernel(const float* __restrict__ z,
                              const float* __restrict__ lng,
                              const float* __restrict__ lnb,
                              const float* __restrict__ Wz,
                              float* __restrict__ S) {
    __shared__ float sWz[CZ * H_];
    __shared__ float sg[CZ], sb[CZ];
    for (int t = threadIdx.x; t < CZ * H_; t += 256) sWz[t] = Wz[t];
    for (int t = threadIdx.x; t < CZ; t += 256) { sg[t] = lng[t]; sb[t] = lnb[t]; }
    __syncthreads();

    const int wave = threadIdx.x >> 5, lane = threadIdx.x & 31;
    const size_t rowId = (size_t)blockIdx.x * 8 + wave;        // i*R + j
    const float4 x = ((const float4*)(z + rowId * CZ))[lane];

    float s = x.x + x.y + x.z + x.w;
    #pragma unroll
    for (int o = 16; o; o >>= 1) s += __shfl_xor(s, o, 32);
    const float mu = s * (1.0f / CZ);
    const float dx = x.x - mu, dy = x.y - mu, dz = x.z - mu, dw = x.w - mu;
    float v = dx*dx + dy*dy + dz*dz + dw*dw;
    #pragma unroll
    for (int o = 16; o; o >>= 1) v += __shfl_xor(v, o, 32);
    const float rinv = rsqrtf(v * (1.0f / CZ) + EPSF);

    const int c0 = lane * 4;
    const float n0 = dx * rinv * sg[c0 + 0] + sb[c0 + 0];
    const float n1 = dy * rinv * sg[c0 + 1] + sb[c0 + 1];
    const float n2 = dz * rinv * sg[c0 + 2] + sb[c0 + 2];
    const float n3 = dw * rinv * sg[c0 + 3] + sb[c0 + 3];

    float acc[H_];
    #pragma unroll
    for (int h = 0; h < H_; ++h)
        acc[h] = n0 * sWz[(c0 + 0) * H_ + h] + n1 * sWz[(c0 + 1) * H_ + h] +
                 n2 * sWz[(c0 + 2) * H_ + h] + n3 * sWz[(c0 + 3) * H_ + h];
    #pragma unroll
    for (int o = 16; o; o >>= 1) {
        #pragma unroll
        for (int h = 0; h < H_; ++h) acc[h] += __shfl_xor(acc[h], o, 32);
    }
    float outv = acc[0];
    #pragma unroll
    for (int h = 1; h < H_; ++h) if (lane == h) outv = acc[h];
    if (lane < H_) S[(size_t)lane * R_ * R_ + rowId] = outv;
}

// ---------------------------------------------------------------------------
// WMMA f16 operand loader: element e of v16h <-> K = 16*(e>>3) + 8*(lane>>4) + (e&7)
// Given a row pointer (row base + 8*hi already applied), two uint4 loads cover it.
// ---------------------------------------------------------------------------
__device__ __forceinline__ v16h load_frag(const _Float16* p) {
    V16H f;
    f.q[0] = *(const uint4*)(p);
    f.q[1] = *(const uint4*)(p + 16);
    return f.v;
}

// ---------------------------------------------------------------------------
// Kernel 4: fused Q/K/V/G projections.  1 wave = one 16x16 tile of one matrix.
// block = 128 thr (4 waves); total waves = 4 * 64 * 32 = 8192 -> 2048 blocks.
// ---------------------------------------------------------------------------
__global__ void proj_kernel(const _Float16* __restrict__ Mh,
                            const _Float16* __restrict__ Wqt, const _Float16* __restrict__ Wkt,
                            const _Float16* __restrict__ Wvt, const _Float16* __restrict__ Wgt,
                            const float* __restrict__ bg,
                            _Float16* __restrict__ Qh, _Float16* __restrict__ Kh,
                            _Float16* __restrict__ Vt, float* __restrict__ G) {
    const int w    = blockIdx.x * 4 + (threadIdx.x >> 5);
    const int lane = threadIdx.x & 31;
    const int mat  = w >> 11;           // 2048 tiles per matrix
    const int t    = w & 2047;
    const int mt   = t >> 5, nt = t & 31;
    const _Float16* W = (mat == 0) ? Wqt : (mat == 1) ? Wkt : (mat == 2) ? Wvt : Wgt;

    const int lo = lane & 15, hi = lane >> 4;
    const _Float16* pa = Mh + (size_t)((mt << 4) + lo) * CIN + 8 * hi;
    const _Float16* pb = W  + (size_t)((nt << 4) + lo) * CIN + 8 * hi;

    v8f c = {};
    #pragma unroll
    for (int k0 = 0; k0 < CIN; k0 += 32) {
        v16h a = load_frag(pa + k0);
        v16h b = load_frag(pb + k0);
        c = __builtin_amdgcn_wmma_f32_16x16x32_f16(false, a, false, b, (short)0, c, false, false);
    }

    const int col = (nt << 4) + lo;
    #pragma unroll
    for (int r = 0; r < 8; ++r) {
        const int row = (mt << 4) + r + 8 * hi;
        const float v = c[r];
        if (mat == 0)      Qh[(size_t)row * CHID + col] = (_Float16)(v * QSCALE);
        else if (mat == 1) Kh[(size_t)row * CHID + col] = (_Float16)v;
        else if (mat == 2) Vt[(size_t)col * R_ + row]   = (_Float16)v;   // [h*64+d][j]
        else               G[(size_t)row * CHID + col]  = 1.0f / (1.0f + __expf(-(v + bg[col])));
    }
}

// ---------------------------------------------------------------------------
// Kernel 5: scores = Q K^T + bias (in place over bias buffer).
// 1 wave per 16x16 tile per head: 8*64*64 = 32768 waves; block 256 -> 4096 blocks.
// ---------------------------------------------------------------------------
__global__ void qk_kernel(const _Float16* __restrict__ Qh,
                          const _Float16* __restrict__ Kh,
                          float* __restrict__ S) {
    const int w    = blockIdx.x * 8 + (threadIdx.x >> 5);
    const int lane = threadIdx.x & 31;
    const int h    = w >> 12;
    const int t    = w & 4095;
    const int it   = t >> 6, jt = t & 63;
    const int lo = lane & 15, hi = lane >> 4;

    const _Float16* pa = Qh + (size_t)((it << 4) + lo) * CHID + (h << 6) + 8 * hi;
    const _Float16* pb = Kh + (size_t)((jt << 4) + lo) * CHID + (h << 6) + 8 * hi;

    v8f c = {};
    #pragma unroll
    for (int k0 = 0; k0 < DH_; k0 += 32) {
        v16h a = load_frag(pa + k0);
        v16h b = load_frag(pb + k0);
        c = __builtin_amdgcn_wmma_f32_16x16x32_f16(false, a, false, b, (short)0, c, false, false);
    }

    const size_t base = (size_t)h * R_ * R_;
    #pragma unroll
    for (int r = 0; r < 8; ++r) {
        const int row = (it << 4) + r + 8 * hi;
        const int col = (jt << 4) + lo;
        const size_t idx = base + (size_t)row * R_ + col;
        S[idx] = c[r] + S[idx];                   // add pair bias
    }
}

// ---------------------------------------------------------------------------
// Kernel 6: softmax over each (h,i) row; emit f16 attention matrix.
// 1 block (256 thr) per row; 8192 blocks.
// ---------------------------------------------------------------------------
__global__ void softmax_kernel(const float* __restrict__ S, _Float16* __restrict__ Ah) {
    const int row = blockIdx.x;            // h*R + i
    const int tid = threadIdx.x;
    const float4 x = ((const float4*)(S + (size_t)row * R_))[tid];

    __shared__ float red[256];
    float mx = fmaxf(fmaxf(x.x, x.y), fmaxf(x.z, x.w));
    red[tid] = mx; __syncthreads();
    #pragma unroll
    for (int o = 128; o > 0; o >>= 1) { if (tid < o) red[tid] = fmaxf(red[tid], red[tid + o]); __syncthreads(); }
    mx = red[0]; __syncthreads();

    const float e0 = __expf(x.x - mx), e1 = __expf(x.y - mx);
    const float e2 = __expf(x.z - mx), e3 = __expf(x.w - mx);
    red[tid] = e0 + e1 + e2 + e3; __syncthreads();
    #pragma unroll
    for (int o = 128; o > 0; o >>= 1) { if (tid < o) red[tid] += red[tid + o]; __syncthreads(); }
    const float inv = 1.0f / red[0];

    H4 o;
    o.h[0] = (_Float16)(e0 * inv); o.h[1] = (_Float16)(e1 * inv);
    o.h[2] = (_Float16)(e2 * inv); o.h[3] = (_Float16)(e3 * inv);
    *(uint2*)(Ah + (size_t)row * R_ + tid * 4) = o.u;
}

// ---------------------------------------------------------------------------
// Kernel 7: O = A V, fused with sigmoid gate -> f16.
// waves = 8 heads * 64 itiles * 4 dtiles = 2048; block 256 -> 256 blocks.
// ---------------------------------------------------------------------------
__global__ void av_kernel(const _Float16* __restrict__ Ah,
                          const _Float16* __restrict__ Vt,
                          const float* __restrict__ G,
                          _Float16* __restrict__ OGh) {
    const int w    = blockIdx.x * 8 + (threadIdx.x >> 5);
    const int lane = threadIdx.x & 31;
    const int h    = w >> 8;
    const int t    = w & 255;
    const int it   = t >> 2, dt = t & 3;
    const int lo = lane & 15, hi = lane >> 4;

    const _Float16* pa = Ah + (size_t)h * R_ * R_ + (size_t)((it << 4) + lo) * R_ + 8 * hi;
    const _Float16* pb = Vt + (size_t)((h << 6) + (dt << 4) + lo) * R_ + 8 * hi;

    v8f c = {};
    #pragma unroll 8
    for (int k0 = 0; k0 < R_; k0 += 32) {
        v16h a = load_frag(pa + k0);
        v16h b = load_frag(pb + k0);
        c = __builtin_amdgcn_wmma_f32_16x16x32_f16(false, a, false, b, (short)0, c, false, false);
    }

    #pragma unroll
    for (int r = 0; r < 8; ++r) {
        const int row = (it << 4) + r + 8 * hi;
        const int col = (h << 6) + (dt << 4) + lo;
        const float val = c[r] * G[(size_t)row * CHID + col];
        OGh[(size_t)row * CHID + col] = (_Float16)val;
    }
}

// ---------------------------------------------------------------------------
// Kernel 8: out = OG @ Wo + bo.  waves = 64*32 = 2048; block 256 -> 256 blocks.
// ---------------------------------------------------------------------------
__global__ void out_kernel(const _Float16* __restrict__ OGh,
                           const _Float16* __restrict__ Wot,
                           const float* __restrict__ bo,
                           float* __restrict__ out) {
    const int w    = blockIdx.x * 8 + (threadIdx.x >> 5);
    const int lane = threadIdx.x & 31;
    const int mt   = w >> 5, nt = w & 31;
    const int lo = lane & 15, hi = lane >> 4;

    const _Float16* pa = OGh + (size_t)((mt << 4) + lo) * CHID + 8 * hi;
    const _Float16* pb = Wot + (size_t)((nt << 4) + lo) * CHID + 8 * hi;

    v8f c = {};
    #pragma unroll
    for (int k0 = 0; k0 < CHID; k0 += 32) {
        v16h a = load_frag(pa + k0);
        v16h b = load_frag(pb + k0);
        c = __builtin_amdgcn_wmma_f32_16x16x32_f16(false, a, false, b, (short)0, c, false, false);
    }

    #pragma unroll
    for (int r = 0; r < 8; ++r) {
        const int row = (mt << 4) + r + 8 * hi;
        const int col = (nt << 4) + lo;
        out[(size_t)row * CIN + col] = c[r] + bo[col];
    }
}

// ---------------------------------------------------------------------------
extern "C" void kernel_launch(void* const* d_in, const int* in_sizes, int n_in,
                              void* d_out, int out_size, void* d_ws, size_t ws_size,
                              hipStream_t stream) {
    const float* m     = (const float*)d_in[0];
    const float* z     = (const float*)d_in[1];
    const float* ln_mg = (const float*)d_in[2];
    const float* ln_mb = (const float*)d_in[3];
    const float* ln_zg = (const float*)d_in[4];
    const float* ln_zb = (const float*)d_in[5];
    const float* Wq    = (const float*)d_in[6];
    const float* Wk    = (const float*)d_in[7];
    const float* Wv    = (const float*)d_in[8];
    const float* Wz    = (const float*)d_in[9];
    const float* Wg    = (const float*)d_in[10];
    const float* bg    = (const float*)d_in[11];
    const float* Wo    = (const float*)d_in[12];
    const float* bo    = (const float*)d_in[13];
    float* outp        = (float*)d_out;

    char* ws = (char*)d_ws;
    size_t off = 0;
    float*    S   = (float*)(ws + off);    off += (size_t)H_ * R_ * R_ * 4;   // bias -> scores
    _Float16* Ah  = (_Float16*)(ws + off); off += (size_t)H_ * R_ * R_ * 2;   // softmax f16
    _Float16* Mh  = (_Float16*)(ws + off); off += (size_t)R_ * CIN * 2;       // LN(m) f16
    _Float16* Wqt = (_Float16*)(ws + off); off += (size_t)512 * 512 * 2;
    _Float16* Wkt = (_Float16*)(ws + off); off += (size_t)512 * 512 * 2;
    _Float16* Wvt = (_Float16*)(ws + off); off += (size_t)512 * 512 * 2;
    _Float16* Wgt = (_Float16*)(ws + off); off += (size_t)512 * 512 * 2;
    _Float16* Wot = (_Float16*)(ws + off); off += (size_t)512 * 512 * 2;
    _Float16* Qh  = (_Float16*)(ws + off); off += (size_t)R_ * CHID * 2;
    _Float16* Kh  = (_Float16*)(ws + off); off += (size_t)R_ * CHID * 2;
    _Float16* Vt  = (_Float16*)(ws + off); off += (size_t)H_ * DH_ * R_ * 2;  // [h*64+d][j]
    float*    G   = (float*)(ws + off);    off += (size_t)R_ * CHID * 4;      // sigmoid gate
    _Float16* OGh = (_Float16*)(ws + off); off += (size_t)R_ * CHID * 2;

    ln_m_kernel   <<<R_, 128, 0, stream>>>(m, ln_mg, ln_mb, Mh);
    cvt_w_kernel  <<<dim3(16, 16, 5), dim3(32, 8), 0, stream>>>(Wq, Wk, Wv, Wg, Wo,
                                                                Wqt, Wkt, Wvt, Wgt, Wot);
    z_bias_kernel <<<(R_ * R_) / 8, 256, 0, stream>>>(z, ln_zg, ln_zb, Wz, S);
    proj_kernel   <<<2048, 128, 0, stream>>>(Mh, Wqt, Wkt, Wvt, Wgt, bg, Qh, Kh, Vt, G);
    qk_kernel     <<<4096, 256, 0, stream>>>(Qh, Kh, S);
    softmax_kernel<<<H_ * R_, 256, 0, stream>>>(S, Ah);
    av_kernel     <<<256, 256, 0, stream>>>(Ah, Vt, G, OGh);
    out_kernel    <<<256, 256, 0, stream>>>(OGh, Wot, bo, outp);
}
